// Deep_Lagrangian_Network_180388626794
// MI455X (gfx1250) — compile-verified, matrix operand order checked by reading, not run on hardware
//
#include <hip/hip_runtime.h>
#include <hip/hip_bf16.h>

typedef __attribute__((ext_vector_type(16))) __bf16 v16bf;
typedef __attribute__((ext_vector_type(8)))  __bf16 v8bf;
typedef __attribute__((ext_vector_type(8)))  float  v8f;

#define N_DOF    7
#define HIDDEN   256
#define SAMPLES  16          // samples per workgroup
#define ROWS     128         // SAMPLES * 8 (primal + 7 tangents)
#define ASTRIDE  272         // padded bf16 row stride (544B: 16B aligned, conflict-free)
#define WSTRIDE  272
#define ZCOLS    48          // 7 g + 7 Ld + 21 Lt, padded to 48

// LDS layout (bytes)
#define OFF_A0   0
#define OFF_A1   (OFF_A0 + ROWS*ASTRIDE*2)          // 69632
#define OFF_WB   (OFF_A1 + ROWS*ASTRIDE*2)          // 139264
#define OFF_Z3   (OFF_WB + 256*WSTRIDE*2)           // 278528
#define OFF_BIAS (OFF_Z3 + ROWS*ZCOLS*4)            // 303104
#define OFF_QS   (OFF_BIAS + 512*4)                 // 305152
#define OFF_LDT  (OFF_QS + SAMPLES*N_DOF*4)         // 305600 : 16*49 f32
#define OFF_CORR (OFF_LDT + SAMPLES*49*4)           // 308736 : 16*8 f32
#define SMEM_BYTES (OFF_CORR + SAMPLES*8*4)         // 309248 < 320KB

// Fused branchless softplus + sigmoid sharing one exp:
//   t  = exp(-|x|)
//   sp = max(x,0) + log(1+t)
//   sg = (x>=0 ? 1 : t) * rcp(1+t)      (v_rcp_f32; plenty for bf16 targets)
__device__ __forceinline__ void sp_sig(float x, float& sp, float& sg) {
    const float t  = __expf(-fabsf(x));
    const float p1 = 1.f + t;
    sp = fmaxf(x, 0.f) + __logf(p1);
    sg = (x >= 0.f ? 1.f : t) * __builtin_amdgcn_rcpf(p1);
}
__device__ __forceinline__ float softplus_f(float x) {
    return fmaxf(x, 0.f) + __logf(1.f + __expf(-fabsf(x)));
}

// Load one 16x32 bf16 fragment (A or B operand) from an LDS row.
// Per ISA layout: lanes 0-15 carry K chunks {k0..k0+7, k0+16..k0+23},
// lanes 16-31 carry {k0+8..k0+15, k0+24..k0+31}; caller folds the +8 into k0.
__device__ __forceinline__ v16bf lds_frag(const __bf16* rowp, int k0) {
    v8bf c0 = *reinterpret_cast<const v8bf*>(rowp + k0);
    v8bf c1 = *reinterpret_cast<const v8bf*>(rowp + k0 + 16);
    return __builtin_shufflevector(c0, c1, 0,1,2,3,4,5,6,7,8,9,10,11,12,13,14,15);
}

// Same fragment but sourced from global f32 weights (heads), with a zero mask.
__device__ __forceinline__ v16bf g32_frag(const float* p, float sc) {
    v16bf r;
#pragma unroll
    for (int i = 0; i < 8; ++i) r[i]     = (__bf16)(p[i]      * sc);
#pragma unroll
    for (int i = 0; i < 8; ++i) r[8 + i] = (__bf16)(p[16 + i] * sc);
    return r;
}

// Cooperative f32 -> bf16 weight staging of a 256x256 matrix into padded LDS.
__device__ __forceinline__ void stage_weights(const float* __restrict__ W,
                                              __bf16* WB, int tid) {
    const int r0 = tid >> 6;             // 0..3
    const int c4 = (tid & 63) * 4;       // 0..252
#pragma unroll 4
    for (int i = 0; i < 64; ++i) {
        const int row = i * 4 + r0;
        const float4 v = *reinterpret_cast<const float4*>(W + row * 256 + c4);
        __bf16* d = WB + row * WSTRIDE + c4;
        d[0] = (__bf16)v.x; d[1] = (__bf16)v.y;
        d[2] = (__bf16)v.z; d[3] = (__bf16)v.w;
    }
}

// One trunk layer: Z = A(128x256) * W^T, softplus on primal rows,
// sigmoid(z_primal)-scaling of the 7 tangent rows (lane-local in C layout).
__device__ __forceinline__ void gemm_layer(const __bf16* Ain, const __bf16* WB,
                                           const float* bias, __bf16* Aout) {
    const int lane = threadIdx.x & 31;
    const int wave = threadIdx.x >> 5;       // row tile 0..7
    const int lrow = lane & 15;
    const int hiK  = (lane >> 4) * 8;        // second-half-of-wave K offset
    const __bf16* arow = Ain + (wave * 16 + lrow) * ASTRIDE;

    v16bf af[8];
#pragma unroll
    for (int kk = 0; kk < 8; ++kk) af[kk] = lds_frag(arow, kk * 32 + hiK);

    const int rbase = wave * 16 + ((lane >> 4) ? 8 : 0);
#pragma unroll 2
    for (int nt = 0; nt < 16; ++nt) {
        v8f acc = {0.f,0.f,0.f,0.f,0.f,0.f,0.f,0.f};
        const __bf16* wrow = WB + (nt * 16 + lrow) * WSTRIDE;
#pragma unroll
        for (int kk = 0; kk < 8; ++kk) {
            v16bf bf = lds_frag(wrow, kk * 32 + hiK);
            acc = __builtin_amdgcn_wmma_f32_16x16x32_bf16(
                      false, af[kk], false, bf, (short)0, acc, false, false);
        }
        const int ncol = nt * 16 + lrow;
        const float zp = acc[0] + bias[ncol];
        float hv, dv;
        sp_sig(zp, hv, dv);
        Aout[(rbase + 0) * ASTRIDE + ncol] = (__bf16)hv;
#pragma unroll
        for (int r = 1; r < 8; ++r)
            Aout[(rbase + r) * ASTRIDE + ncol] = (__bf16)(dv * acc[r]);
    }
}

extern "C" __global__ __launch_bounds__(256)
void delan_fused_kernel(const float* __restrict__ q,  const float* __restrict__ qd,
                        const float* __restrict__ qdd,
                        const float* __restrict__ W0, const float* __restrict__ b0,
                        const float* __restrict__ W1, const float* __restrict__ b1,
                        const float* __restrict__ W2, const float* __restrict__ b2,
                        const float* __restrict__ Wg, const float* __restrict__ bg,
                        const float* __restrict__ Wld, const float* __restrict__ bld,
                        const float* __restrict__ Wlt, const float* __restrict__ blt,
                        const float* __restrict__ fd, const float* __restrict__ fc,
                        const float* __restrict__ fs, const float* __restrict__ fv,
                        float* __restrict__ out, int B) {
    extern __shared__ char smem[];
    __bf16* A0  = (__bf16*)(smem + OFF_A0);
    __bf16* A1  = (__bf16*)(smem + OFF_A1);
    __bf16* WB  = (__bf16*)(smem + OFF_WB);
    float*  Z3  = (float*)(smem + OFF_Z3);
    float*  BS  = (float*)(smem + OFF_BIAS);
    float*  QS  = (float*)(smem + OFF_QS);
    float*  LDT = (float*)(smem + OFF_LDT);    // per-sample L_dt accumulators
    float*  CORR= (float*)(smem + OFF_CORR);   // per-sample 0.5*qd^T H_dq qd

    const int tid = threadIdx.x;
    const int s0  = blockIdx.x * SAMPLES;

    // ---- stage q, biases, W1 ----
    if (tid < SAMPLES * N_DOF) {
        const long long gi = (long long)s0 * N_DOF + tid;
        QS[tid] = (gi < (long long)B * N_DOF) ? q[gi] : 0.f;
    }
    BS[tid]       = b1[tid];
    BS[256 + tid] = b2[tid];
    stage_weights(W1, WB, tid);
    __builtin_prefetch(W2 + tid * 256, 0, 1);   // warm W2 while layer 0/1 run
    __syncthreads();

    // ---- layer 0: z0 = W0 q + b0; primal + 7 tangent rows into A0 ----
    {
        const int h = tid;
        float w0r[N_DOF];
#pragma unroll
        for (int j = 0; j < N_DOF; ++j) w0r[j] = W0[h * N_DOF + j];
        const float bb = b0[h];
#pragma unroll 1
        for (int s = 0; s < SAMPLES; ++s) {
            float z = bb;
#pragma unroll
            for (int j = 0; j < N_DOF; ++j) z += w0r[j] * QS[s * N_DOF + j];
            float hv, dv;
            sp_sig(z, hv, dv);
            A0[(s * 8) * ASTRIDE + h] = (__bf16)hv;
#pragma unroll
            for (int j = 0; j < N_DOF; ++j)
                A0[(s * 8 + 1 + j) * ASTRIDE + h] = (__bf16)(dv * w0r[j]);
        }
    }
    __syncthreads();

    // ---- layer 1 / layer 2 (WMMA) ----
    gemm_layer(A0, WB, BS, A1);
    __syncthreads();
    stage_weights(W2, WB, tid);
    __syncthreads();
    gemm_layer(A1, WB, BS + 256, A0);
    __syncthreads();

    // zero L_dt / CORR accumulators (synced by the post-head barrier)
    for (int i = tid; i < SAMPLES * 49 + SAMPLES * 8; i += 256) LDT[i] = 0.f;

    // ---- heads: [128 x 256] x [256 x 48] -> Z3 (raw, pre-bias) ----
    {
        const int lane = tid & 31;
        const int wave = tid >> 5;
        const int lrow = lane & 15;
        const int hiK  = (lane >> 4) * 8;
        const __bf16* arow = A0 + (wave * 16 + lrow) * ASTRIDE;
        v16bf af[8];
#pragma unroll
        for (int kk = 0; kk < 8; ++kk) af[kk] = lds_frag(arow, kk * 32 + hiK);
        const int rbase = wave * 16 + ((lane >> 4) ? 8 : 0);
#pragma unroll 1
        for (int nt = 0; nt < 3; ++nt) {
            const int ncol = nt * 16 + lrow;
            const float* hw = Wg;
            float sc = 0.f;
            if (ncol < 7)       { hw = Wg  + ncol * 256;        sc = 1.f; }
            else if (ncol < 14) { hw = Wld + (ncol - 7) * 256;  sc = 1.f; }
            else if (ncol < 35) { hw = Wlt + (ncol - 14) * 256; sc = 1.f; }
            v8f acc = {0.f,0.f,0.f,0.f,0.f,0.f,0.f,0.f};
#pragma unroll
            for (int kk = 0; kk < 8; ++kk) {
                v16bf bf = g32_frag(hw + kk * 32 + hiK, sc);
                acc = __builtin_amdgcn_wmma_f32_16x16x32_bf16(
                          false, af[kk], false, bf, (short)0, acc, false, false);
            }
#pragma unroll
            for (int r = 0; r < 8; ++r)
                Z3[(rbase + r) * ZCOLS + ncol] = acc[r];
        }
    }
    __syncthreads();

    // ---- physics phase 1 : thread = (sample, tangent c) ; 112 lanes ----
    // Builds L_dt via LDS float atomics and the quadratic Coriolis correction.
    if (tid < SAMPLES * N_DOF) {
        const int s  = tid / N_DOF;
        const int c2 = tid % N_DOF;
        const long long gs = (long long)s0 + s;
        if (gs < B) {
            const float* zp = Z3 + (s * 8) * ZCOLS;
            const float* zt = Z3 + (s * 8 + 1 + c2) * ZCOLS;
            float qdv[7];
#pragma unroll
            for (int i = 0; i < 7; ++i) qdv[i] = qd[gs*7+i];

            float sgd[7], L[7][7];
#pragma unroll
            for (int i = 0; i < 7; ++i)
#pragma unroll
                for (int j = 0; j < 7; ++j) L[i][j] = 0.f;
#pragma unroll
            for (int i = 0; i < 7; ++i) {
                const float ldr = zp[7 + i] + bld[i];
                float sp, sg;
                sp_sig(ldr, sp, sg);
                sgd[i]  = sg;
                L[i][i] = sp + 1e-3f;
            }
            {
                int m = 0;
#pragma unroll
                for (int r = 1; r < 7; ++r)
#pragma unroll
                    for (int cc = 0; cc < r; ++cc) { L[r][cc] = zp[14 + m] + blt[m]; ++m; }
            }

            // L_dq(:,:,c2)
            float Ldq[7][7];
#pragma unroll
            for (int i = 0; i < 7; ++i)
#pragma unroll
                for (int j = 0; j < 7; ++j) Ldq[i][j] = 0.f;
#pragma unroll
            for (int i = 0; i < 7; ++i) Ldq[i][i] = sgd[i] * zt[7 + i];
            {
                int m = 0;
#pragma unroll
                for (int r = 1; r < 7; ++r)
#pragma unroll
                    for (int cc = 0; cc < r; ++cc) { Ldq[r][cc] = zt[14 + m]; ++m; }
            }

            // accumulate L_dt += L_dq(:,:,c2) * qd[c2]   (ds_add_f32)
            {
                const float w = qdv[c2];
                float* ldts = LDT + s * 49;
#pragma unroll
                for (int i = 0; i < 7; ++i)
#pragma unroll
                    for (int j = 0; j <= i; ++j)
                        atomicAdd(&ldts[i * 7 + j], Ldq[i][j] * w);
            }

            // u = qd^T L ; v = qd^T L_dq ; accq = qd^T H_dq(:,:,c2) qd
            float u[7], v[7];
#pragma unroll
            for (int j = 0; j < 7; ++j) {
                float a = 0.f, b = 0.f;
#pragma unroll
                for (int k = 0; k < 7; ++k) { a += qdv[k] * L[k][j]; b += qdv[k] * Ldq[k][j]; }
                u[j] = a; v[j] = b;
            }
            float accq = 0.f;
#pragma unroll
            for (int i = 0; i < 7; ++i) {
                float hq = 0.f;
#pragma unroll
                for (int j = 0; j < 7; ++j) hq += Ldq[i][j] * u[j] + L[i][j] * v[j];
                accq += qdv[i] * hq;
            }
            CORR[s * 8 + c2] = 0.5f * accq;
        }
    }
    __syncthreads();

    // ---- physics phase 2 : thread = (sample, output row i) ; 112 lanes ----
    if (tid < SAMPLES * N_DOF) {
        const int s = tid / N_DOF;
        const int i = tid % N_DOF;
        const long long gs = (long long)s0 + s;
        if (gs < B) {
            const float* zp = Z3 + (s * 8) * ZCOLS;
            float qdv[7], qddv[7];
#pragma unroll
            for (int k = 0; k < 7; ++k) { qdv[k] = qd[gs*7+k]; qddv[k] = qdd[gs*7+k]; }

            float L[7][7];
#pragma unroll
            for (int r = 0; r < 7; ++r)
#pragma unroll
                for (int j = 0; j < 7; ++j) L[r][j] = 0.f;
            float Ldraw_i = 0.f;
#pragma unroll
            for (int r = 0; r < 7; ++r) {
                const float ldr = zp[7 + r] + bld[r];
                if (r == i) Ldraw_i = ldr;
                L[r][r] = softplus_f(ldr) + 1e-3f;
            }
            {
                int m = 0;
#pragma unroll
                for (int r = 1; r < 7; ++r)
#pragma unroll
                    for (int cc = 0; cc < r; ++cc) { L[r][cc] = zp[14 + m] + blt[m]; ++m; }
            }

            const float* ldts = LDT + s * 49;
            float Ldt[7][7];
#pragma unroll
            for (int r = 0; r < 7; ++r)
#pragma unroll
                for (int j = 0; j < 7; ++j) Ldt[r][j] = ldts[r * 7 + j];

            // Hm row i, Coriolis component i
            float Hrow[7];
            float ci = 0.f;
#pragma unroll
            for (int k = 0; k < 7; ++k) {
                float a = 0.f, h = 0.f;
#pragma unroll
                for (int j = 0; j < 7; ++j) {
                    a += L[i][j] * L[k][j];
                    h += L[i][j] * Ldt[k][j] + Ldt[i][j] * L[k][j];
                }
                Hrow[k] = a;
                ci += h * qdv[k];
            }
            ci -= CORR[s * 8 + i];

            const float gv  = zp[i] + bg[i];
            const float fvc = fmaxf(fv[i], 1e-3f);
            const float fr  = (fc[i] + fs[i] * __expf(-qdv[i]*qdv[i] / fvc))
                              * tanhf(100.f * qdv[i]) + fd[i] * qdv[i];
            float tau = ci + gv + fr;
#pragma unroll
            for (int k = 0; k < 7; ++k) tau += Hrow[k] * qddv[k];

            const long long Bl = B;
            out[gs*7 + i]            = tau;       // tau_pred
            out[56*Bl + gs*7 + i]    = ci;        // c
            out[63*Bl + gs*7 + i]    = gv;        // g
            out[70*Bl + gs*7 + i]    = fr;        // tau_fric
            out[77*Bl + gs*7 + i]    = Ldraw_i;   // Ld_raw
            float* Hm_o = out + 7*Bl + gs*49 + (long long)i*7;
#pragma unroll
            for (int k = 0; k < 7; ++k) Hm_o[k] = Hrow[k];
        }
    }
}

extern "C" void kernel_launch(void* const* d_in, const int* in_sizes, int n_in,
                              void* d_out, int out_size, void* d_ws, size_t ws_size,
                              hipStream_t stream) {
    (void)n_in; (void)d_ws; (void)ws_size; (void)out_size;
    const float* q   = (const float*)d_in[0];
    const float* qd  = (const float*)d_in[1];
    const float* qdd = (const float*)d_in[2];
    const float* W0  = (const float*)d_in[3];
    const float* b0  = (const float*)d_in[4];
    const float* W1  = (const float*)d_in[5];
    const float* b1  = (const float*)d_in[6];
    const float* W2  = (const float*)d_in[7];
    const float* b2  = (const float*)d_in[8];
    const float* Wg  = (const float*)d_in[9];
    const float* bg  = (const float*)d_in[10];
    const float* Wld = (const float*)d_in[11];
    const float* bld = (const float*)d_in[12];
    const float* Wlt = (const float*)d_in[13];
    const float* blt = (const float*)d_in[14];
    const float* fd  = (const float*)d_in[15];
    const float* fc  = (const float*)d_in[16];
    const float* fs  = (const float*)d_in[17];
    const float* fv  = (const float*)d_in[18];

    const int B = in_sizes[0] / N_DOF;
    const int grid = (B + SAMPLES - 1) / SAMPLES;

    (void)hipFuncSetAttribute((const void*)delan_fused_kernel,
                              hipFuncAttributeMaxDynamicSharedMemorySize,
                              SMEM_BYTES);

    delan_fused_kernel<<<grid, 256, SMEM_BYTES, stream>>>(
        q, qd, qdd, W0, b0, W1, b1, W2, b2, Wg, bg, Wld, bld, Wlt, blt,
        fd, fc, fs, fv, (float*)d_out, B);
}
